// D3_Net_Memory_28879360098416
// MI455X (gfx1250) — compile-verified, matrix operand order checked by reading, not run on hardware
//
#include <hip/hip_runtime.h>
#include <hip/hip_bf16.h>

typedef __attribute__((ext_vector_type(16))) _Float16 v16h;
typedef __attribute__((ext_vector_type(2)))  __fp16   pk2h;   // cvt_pkrtz result type
typedef __attribute__((ext_vector_type(8)))  float    v8f;

constexpr int NB   = 2048;   // batch
constexpr int KD   = 2048;   // codebook size
constexpr int CDIM = 400;    // feature dim

// ---------------------------------------------------------------------------
// Stage 1: conv1(valid,relu) -> pool1(stride-2 depth conv) -> conv2(relu)
// One workgroup per sample; whole intermediate chain lives in LDS (234 KB).
// ---------------------------------------------------------------------------
__global__ void conv_stage1(const float* __restrict__ x,
                            const float* __restrict__ w1, const float* __restrict__ b1,
                            const float* __restrict__ wp, const float* __restrict__ bp,
                            const float* __restrict__ w2, const float* __restrict__ b2,
                            float* __restrict__ h2) {
    extern __shared__ float smem[];
    float* xs = smem;                 // 1 x 200 x 5 x 5 = 5000
    float* c1 = smem + 5000;          // 20 x 198 x 9   = 35640
    float* p1 = c1 + 35640;           // 20 x 99 x 9    = 17820
    const int s = blockIdx.x, tid = threadIdx.x;
    const float* xin = x + (size_t)s * 5000;
    for (int i = tid; i < 5000; i += 256) xs[i] = xin[i];
    __syncthreads();
    // conv1 + relu: out (20,198,3,3)
    for (int idx = tid; idx < 20 * 198 * 9; idx += 256) {
        int oc = idx / (198 * 9), rem = idx % (198 * 9);
        int d = rem / 9, hw = rem % 9, h = hw / 3, w = hw % 3;
        float acc = b1[oc];
        const float* wk = w1 + oc * 27;
        #pragma unroll
        for (int kd = 0; kd < 3; ++kd)
            #pragma unroll
            for (int kh = 0; kh < 3; ++kh)
                #pragma unroll
                for (int kw = 0; kw < 3; ++kw)
                    acc += wk[kd * 9 + kh * 3 + kw] * xs[(d + kd) * 25 + (h + kh) * 5 + (w + kw)];
        c1[idx] = fmaxf(acc, 0.0f);
    }
    __syncthreads();
    // pool1: depth conv k=3 stride 2 pad 1 -> (20,99,3,3)
    for (int idx = tid; idx < 20 * 99 * 9; idx += 256) {
        int oc = idx / (99 * 9), rem = idx % (99 * 9);
        int od = rem / 9, hw = rem % 9;
        float acc = bp[oc];
        for (int ic = 0; ic < 20; ++ic)
            #pragma unroll
            for (int kd = 0; kd < 3; ++kd) {
                int dd = 2 * od - 1 + kd;
                if (dd >= 0 && dd < 198)
                    acc += wp[oc * 60 + ic * 3 + kd] * c1[ic * (198 * 9) + dd * 9 + hw];
            }
        p1[idx] = acc;
    }
    __syncthreads();
    // conv2 + relu: (35,99,1,1)
    for (int idx = tid; idx < 35 * 99; idx += 256) {
        int oc = idx / 99, d = idx % 99;
        float acc = b2[oc];
        for (int ic = 0; ic < 20; ++ic)
            #pragma unroll
            for (int kd = 0; kd < 3; ++kd) {
                int dd = d - 1 + kd;
                if (dd < 0 || dd >= 99) continue;
                const float* ww = w2 + oc * 540 + ic * 27 + kd * 9;
                const float* pv = p1 + ic * (99 * 9) + dd * 9;
                #pragma unroll
                for (int t = 0; t < 9; ++t) acc += ww[t] * pv[t];
            }
        h2[(size_t)s * (35 * 99) + idx] = fmaxf(acc, 0.0f);
    }
}

// ---------------------------------------------------------------------------
// Stage 2: pool2 -> conv3(relu) -> conv4(stride-2, relu) -> flatten(910)
// LDS fill uses CDNA5 async global->LDS copy (ASYNCcnt).
// ---------------------------------------------------------------------------
__global__ void conv_stage2(const float* __restrict__ h2,
                            const float* __restrict__ wp2, const float* __restrict__ bp2,
                            const float* __restrict__ w3, const float* __restrict__ b3,
                            const float* __restrict__ w4, const float* __restrict__ b4,
                            float* __restrict__ feat) {
    __shared__ float hh[35 * 99];
    __shared__ float pp[35 * 50];
    __shared__ float c3[35 * 50];
    const int s = blockIdx.x, tid = threadIdx.x;
    const float* src = h2 + (size_t)s * (35 * 99);
    // 3465 floats = 866 x b128 chunks + 1 tail float
    for (int i = tid; i < 866; i += 256) {
        unsigned int lds = (unsigned int)(uintptr_t)&hh[i * 4];
        unsigned int gof = (unsigned int)(i * 16);
        asm volatile("global_load_async_to_lds_b128 %0, %1, %2"
                     :: "v"(lds), "v"(gof), "s"(src) : "memory");
    }
    if (tid == 0) hh[3464] = src[3464];
    asm volatile("s_wait_asynccnt 0x0" ::: "memory");
    __syncthreads();
    for (int idx = tid; idx < 35 * 50; idx += 256) {     // pool2
        int oc = idx / 50, od = idx % 50;
        float acc = bp2[oc];
        for (int ic = 0; ic < 35; ++ic)
            #pragma unroll
            for (int kd = 0; kd < 3; ++kd) {
                int dd = 2 * od - 1 + kd;
                if (dd >= 0 && dd < 99) acc += wp2[oc * 105 + ic * 3 + kd] * hh[ic * 99 + dd];
            }
        pp[idx] = acc;
    }
    __syncthreads();
    for (int idx = tid; idx < 35 * 50; idx += 256) {     // conv3 + relu
        int oc = idx / 50, d = idx % 50;
        float acc = b3[oc];
        for (int ic = 0; ic < 35; ++ic)
            #pragma unroll
            for (int kd = 0; kd < 3; ++kd) {
                int dd = d - 1 + kd;
                if (dd >= 0 && dd < 50) acc += w3[oc * 105 + ic * 3 + kd] * pp[ic * 50 + dd];
            }
        c3[idx] = fmaxf(acc, 0.0f);
    }
    __syncthreads();
    for (int idx = tid; idx < 35 * 26; idx += 256) {     // conv4 stride2 + relu
        int oc = idx / 26, od = idx % 26;
        float acc = b4[oc];
        for (int ic = 0; ic < 35; ++ic)
            #pragma unroll
            for (int kd = 0; kd < 2; ++kd) {
                int dd = 2 * od - 1 + kd;
                if (dd >= 0 && dd < 50) acc += w4[oc * 70 + ic * 2 + kd] * c3[ic * 50 + dd];
            }
        feat[(size_t)s * 910 + idx] = fmaxf(acc, 0.0f);
    }
}

// ---------------------------------------------------------------------------
// TN GEMM via v_wmma_f32_16x16x32_f16:  C[M,N] = A[M,K] . B[N,K]^T (+ bias)
// 256 threads = 8 wave32s; block tile 64x64; TK=64 -> 4 wmma per wave per
// staging round. Staging is branchless: OOB M/N rows are address-clamped
// (their products land in unstored C entries); only the K tail is zeroed
// via select. Two floats per v_cvt_pk_rtz + one b32 LDS store.
// ---------------------------------------------------------------------------
__global__ void gemm_tn_wmma(const float* __restrict__ A, int lda,
                             const float* __restrict__ Bm, int ldb,
                             const float* __restrict__ bias,
                             float* __restrict__ C, int ldc,
                             int M, int N, int K) {
    constexpr int TK = 64;
    __shared__ _Float16 As[64][TK + 8];   // pitch 72 halves = 144 B (16B-aligned rows)
    __shared__ _Float16 Bs[64][TK + 8];
    const int tid = threadIdx.x;
    const int lane = tid & 31, wave = tid >> 5;
    const int wm = wave >> 1, wn = wave & 1;
    const int m0 = blockIdx.y * 64, n0 = blockIdx.x * 64;
    const int half = lane >> 4, lr = lane & 15;
    v8f acc0 = {}, acc1 = {};
    for (int kb = 0; kb < K; kb += TK) {
        // stage A and B 64x64 tiles as f16; 8 k-pairs per thread each
        for (int p = tid; p < 64 * (TK / 2); p += 256) {
            int r  = p >> 5;               // tile row 0..63
            int k2 = (p & 31) << 1;        // even k
            int gk0 = kb + k2, gk1 = gk0 + 1;
            int ck0 = gk0 < K ? gk0 : K - 1;
            int ck1 = gk1 < K ? gk1 : K - 1;

            int gm = m0 + r; gm = gm < M ? gm : M - 1;
            const float* ap = A + (size_t)gm * lda;
            float a0 = ap[ck0], a1 = ap[ck1];
            a0 = gk0 < K ? a0 : 0.0f;
            a1 = gk1 < K ? a1 : 0.0f;
            *(pk2h*)&As[r][k2] = __builtin_amdgcn_cvt_pkrtz(a0, a1);

            int gn = n0 + r; gn = gn < N ? gn : N - 1;
            const float* bp = Bm + (size_t)gn * ldb;
            float b0 = bp[ck0], b1 = bp[ck1];
            b0 = gk0 < K ? b0 : 0.0f;
            b1 = gk1 < K ? b1 : 0.0f;
            *(pk2h*)&Bs[r][k2] = __builtin_amdgcn_cvt_pkrtz(b0, b1);
        }
        if (kb + TK < K) {                 // global_prefetch_b8 of next slab
            int gm = m0 + (tid >> 2); gm = gm < M ? gm : M - 1;
            __builtin_prefetch(&A[(size_t)gm * lda + kb + TK], 0, 0);
        }
        __syncthreads();
        const int row = wm * 16 + lr;
        const int c0  = wn * 32 + lr, c1 = c0 + 16;
        #pragma unroll
        for (int kk = 0; kk < TK; kk += 32) {
            v16h af, bf0, bf1;
            #pragma unroll
            for (int e = 0; e < 8; ++e) {  // 16B-contiguous runs -> ds_load_b128
                af[e]      = As[row][kk + half * 8 + e];
                af[8 + e]  = As[row][kk + 16 + half * 8 + e];
                bf0[e]     = Bs[c0][kk + half * 8 + e];
                bf0[8 + e] = Bs[c0][kk + 16 + half * 8 + e];
                bf1[e]     = Bs[c1][kk + half * 8 + e];
                bf1[8 + e] = Bs[c1][kk + 16 + half * 8 + e];
            }
            acc0 = __builtin_amdgcn_wmma_f32_16x16x32_f16(false, af, false, bf0,
                                                          (short)0, acc0, false, false);
            acc1 = __builtin_amdgcn_wmma_f32_16x16x32_f16(false, af, false, bf1,
                                                          (short)0, acc1, false, false);
        }
        __syncthreads();
    }
    // C/D layout: lanes 0-15 -> M=r, lanes 16-31 -> M=r+8; N = lane&15
    const int colA = n0 + wn * 32 + lr;
    const int colB = colA + 16;
    const float bva = (bias && colA < N) ? bias[colA] : 0.0f;
    const float bvb = (bias && colB < N) ? bias[colB] : 0.0f;
    #pragma unroll
    for (int r = 0; r < 8; ++r) {
        int row = m0 + wm * 16 + r + half * 8;
        if (row < M) {
            if (colA < N) C[(size_t)row * ldc + colA] = acc0[r] + bva;
            if (colB < N) C[(size_t)row * ldc + colB] = acc1[r] + bvb;
        }
    }
}

// ---------------------------------------------------------------------------
// Small helper kernels
// ---------------------------------------------------------------------------
__global__ void l2norm_rows(const float* __restrict__ in, int n,
                            float* __restrict__ out, float* __restrict__ norms) {
    __shared__ float red[256];
    const int row = blockIdx.x, tid = threadIdx.x;
    const float* r = in + (size_t)row * n;
    float s = 0.f;
    for (int i = tid; i < n; i += 256) { float v = r[i]; s += v * v; }
    red[tid] = s; __syncthreads();
    for (int st = 128; st > 0; st >>= 1) { if (tid < st) red[tid] += red[tid + st]; __syncthreads(); }
    float nrm = sqrtf(red[0]);
    float inv = 1.0f / fmaxf(nrm, 1e-12f);
    for (int i = tid; i < n; i += 256) out[(size_t)row * n + i] = r[i] * inv;
    if (tid == 0 && norms) norms[row] = nrm;
}

__global__ void argmax_rows(const float* __restrict__ score, int n, int* __restrict__ ind) {
    __shared__ float bv[256]; __shared__ int bi[256];
    const int row = blockIdx.x, tid = threadIdx.x;
    const float* r = score + (size_t)row * n;
    float best = -3.4e38f; int besti = 0;
    for (int i = tid; i < n; i += 256) { float v = r[i]; if (v > best) { best = v; besti = i; } }
    bv[tid] = best; bi[tid] = besti; __syncthreads();
    for (int st = 128; st > 0; st >>= 1) {
        if (tid < st && (bv[tid + st] > bv[tid] ||
                         (bv[tid + st] == bv[tid] && bi[tid + st] < bi[tid]))) {
            bv[tid] = bv[tid + st]; bi[tid] = bi[tid + st];
        }
        __syncthreads();
    }
    if (tid == 0) ind[row] = bi[0];
}

__global__ void zero_buf(float* p, int n) {
    int i = blockIdx.x * 256 + threadIdx.x;
    if (i < n) p[i] = 0.0f;
}

__global__ void scatter_ema(const float* __restrict__ xx, const int* __restrict__ ind,
                            float* __restrict__ esum, float* __restrict__ counts) {
    const int n = blockIdx.x, tid = threadIdx.x;
    const int k = ind[n];
    for (int c = tid; c < CDIM; c += 256)
        atomicAdd(&esum[(size_t)k * CDIM + c], xx[(size_t)n * CDIM + c]);
    if (tid == 0) atomicAdd(&counts[k], 1.0f);
}

__global__ void col_mean_sum(const float* __restrict__ esum, const float* __restrict__ counts,
                             float* __restrict__ ss) {
    __shared__ float red[256];
    const int c = blockIdx.x, tid = threadIdx.x;
    float s = 0.f;
    for (int k = tid; k < KD; k += 256)
        s += esum[(size_t)k * CDIM + c] / (counts[k] + 1e-6f);
    red[tid] = s; __syncthreads();
    for (int st = 128; st > 0; st >>= 1) { if (tid < st) red[tid] += red[tid + st]; __syncthreads(); }
    if (tid == 0) ss[c] = red[0];
}

__global__ void ema_update(const float* __restrict__ m, const float* __restrict__ esum,
                           const float* __restrict__ counts, const float* __restrict__ ss,
                           float* __restrict__ mnew) {
    int i = blockIdx.x * 256 + threadIdx.x;
    if (i >= KD * CDIM) return;
    int k = i / CDIM, c = i % CDIM;
    float em = esum[i] / (counts[k] + 1e-6f);
    float pd = (em - ss[c]) * (2.0f / ((float)KD * (float)(KD - 1)));
    mnew[i] = m[i] * 0.99f + (em + pd) * 0.01f;
}

__global__ void transpose_kc(const float* __restrict__ m, float* __restrict__ mt) {
    int i = blockIdx.x * 256 + threadIdx.x;
    if (i >= KD * CDIM) return;
    int k = i / CDIM, c = i % CDIM;
    mt[(size_t)c * KD + k] = m[i];
}

__global__ void row_rescale(float* __restrict__ out, const float* __restrict__ xnorm) {
    __shared__ float red[256];
    const int row = blockIdx.x, tid = threadIdx.x;
    float* r = out + (size_t)row * CDIM;
    float s = 0.f;
    for (int i = tid; i < CDIM; i += 256) { float v = r[i]; s += v * v; }
    red[tid] = s; __syncthreads();
    for (int st = 128; st > 0; st >>= 1) { if (tid < st) red[tid] += red[tid + st]; __syncthreads(); }
    float param = xnorm[row] / sqrtf(red[0]);
    for (int i = tid; i < CDIM; i += 256) r[i] *= param;
}

__global__ void sub_ew(const float* __restrict__ a, const float* __restrict__ b,
                       float* __restrict__ o, int n) {
    int i = blockIdx.x * 256 + threadIdx.x;
    if (i < n) o[i] = a[i] - b[i];
}

// ---------------------------------------------------------------------------
extern "C" void kernel_launch(void* const* d_in, const int* in_sizes, int n_in,
                              void* d_out, int out_size, void* d_ws, size_t ws_size,
                              hipStream_t stream) {
    (void)in_sizes; (void)n_in; (void)out_size; (void)ws_size;
    const float* x   = (const float*)d_in[0];
    const float* c1w = (const float*)d_in[1];  const float* c1b = (const float*)d_in[2];
    const float* p1w = (const float*)d_in[3];  const float* p1b = (const float*)d_in[4];
    const float* c2w = (const float*)d_in[5];  const float* c2b = (const float*)d_in[6];
    const float* p2w = (const float*)d_in[7];  const float* p2b = (const float*)d_in[8];
    const float* c3w = (const float*)d_in[9];  const float* c3b = (const float*)d_in[10];
    const float* c4w = (const float*)d_in[11]; const float* c4b = (const float*)d_in[12];
    const float* fw  = (const float*)d_in[13]; const float* fb  = (const float*)d_in[14];
    const float* f1w = (const float*)d_in[15]; const float* f1b = (const float*)d_in[16];
    const float* mem = (const float*)d_in[17];
    const float* uw  = (const float*)d_in[18]; const float* ub  = (const float*)d_in[19];
    const float* cw  = (const float*)d_in[20]; const float* cb  = (const float*)d_in[21];
    float* out = (float*)d_out;

    float* ws = (float*)d_ws;
    size_t off = 0;
    auto alloc = [&](size_t n) { float* p = ws + off; off += n; return p; };
    float* h2     = alloc((size_t)NB * 35 * 99);
    float* featin = alloc((size_t)NB * 910);
    float* fea    = alloc((size_t)NB * CDIM);
    float* xx     = alloc((size_t)NB * CDIM);
    float* xn     = alloc((size_t)NB * CDIM);
    float* xnorm  = alloc(NB);
    float* mn     = alloc((size_t)KD * CDIM);
    float* score  = alloc((size_t)NB * KD);
    float* esum   = alloc((size_t)KD * CDIM);
    float* counts = alloc(KD);
    float* ssb    = alloc(CDIM);
    float* mnew   = alloc((size_t)KD * CDIM);
    float* mnnew  = alloc((size_t)KD * CDIM);
    float* mT     = alloc((size_t)CDIM * KD);
    float* outb   = alloc((size_t)NB * CDIM);
    float* upb    = alloc((size_t)NB * CDIM);
    float* ofeat  = alloc((size_t)NB * CDIM);
    int*   ind    = (int*)alloc(NB);

    // CNN front-end, fused per-sample in LDS
    conv_stage1<<<NB, 256, 58460 * sizeof(float), stream>>>(x, c1w, c1b, p1w, p1b, c2w, c2b, h2);
    conv_stage2<<<NB, 256, 0, stream>>>(h2, p2w, p2b, c3w, c3b, c4w, c4b, featin);

    auto gemm = [&](const float* A, int lda, const float* Bm, int ldb, const float* bias,
                    float* C, int ldc, int M, int N, int K) {
        dim3 g((N + 63) / 64, (M + 63) / 64);
        gemm_tn_wmma<<<g, 256, 0, stream>>>(A, lda, Bm, ldb, bias, C, ldc, M, N, K);
    };

    gemm(featin, 910, fw, 910, fb, fea, CDIM, NB, CDIM, 910);        // feat
    gemm(fea, CDIM, f1w, CDIM, f1b, xx, CDIM, NB, CDIM, CDIM);       // feat1
    l2norm_rows<<<NB, 256, 0, stream>>>(xx, CDIM, xn, xnorm);
    l2norm_rows<<<KD, 256, 0, stream>>>(mem, CDIM, mn, nullptr);
    gemm(xn, CDIM, mn, CDIM, nullptr, score, KD, NB, KD, CDIM);      // score (pre-update)
    argmax_rows<<<NB, 256, 0, stream>>>(score, KD, ind);
    zero_buf<<<(KD * CDIM + 255) / 256, 256, 0, stream>>>(esum, KD * CDIM);
    zero_buf<<<(KD + 255) / 256, 256, 0, stream>>>(counts, KD);
    scatter_ema<<<NB, 256, 0, stream>>>(xx, ind, esum, counts);
    col_mean_sum<<<CDIM, 256, 0, stream>>>(esum, counts, ssb);
    ema_update<<<(KD * CDIM + 255) / 256, 256, 0, stream>>>(mem, esum, counts, ssb, mnew);
    l2norm_rows<<<KD, 256, 0, stream>>>(mnew, CDIM, mnnew, nullptr);
    gemm(xn, CDIM, mnnew, CDIM, nullptr, score, KD, NB, KD, CDIM);   // score (post-update)
    transpose_kc<<<(KD * CDIM + 255) / 256, 256, 0, stream>>>(mnew, mT);
    gemm(score, KD, mT, KD, nullptr, outb, CDIM, NB, CDIM, KD);      // score @ m_new
    row_rescale<<<NB, 256, 0, stream>>>(outb, xnorm);                // (param*score)@m == param*out
    gemm(outb, CDIM, uw, CDIM, ub, upb, CDIM, NB, CDIM, CDIM);       // up
    sub_ew<<<(NB * CDIM + 255) / 256, 256, 0, stream>>>(fea, upb, ofeat, NB * CDIM);
    gemm(ofeat, CDIM, cw, CDIM, cb, out, 16, NB, 16, CDIM);          // cls -> logits
}
